// BiQRNN3DLayer_10737418240001
// MI455X (gfx1250) — matrix operand order repaired and checked by reading, not compiled, not run on hardware
//
#include <hip/hip_runtime.h>

typedef __attribute__((ext_vector_type(16))) _Float16 v16h;
typedef __attribute__((ext_vector_type(8)))  _Float16 v8h;
typedef __attribute__((ext_vector_type(8)))  float    v8f;
typedef int v4i __attribute__((vector_size(16)));

#define HID    16
#define D_DIM  31
#define H_DIM  256
#define W_DIM  256
#define BATCH  2

#define KCH       14                    // K chunks of 32 (27 taps * 16 cin, padded to 28 taps = 448)
#define A_HALFS   (3 * KCH * 32 * 16)   // 21504 f16 (3 M-tiles of pre-swizzled A fragments)
#define A_BYTES   (A_HALFS * 2)         // 43008 B

// padded channel-last f16 copy of x: [b][D+2][H+2][W+2][16]
#define DP 33
#define HP 258
#define WP 258
#define XT_OFF_BYTES 43520              // A region first, then xt (512-aligned)

// ---- dynamic LDS layout (bytes) ----
#define SLOT_HALFS (6 * 18 * 16)                  // 1728 halfs per slab slot
#define SLAB_OFF   A_BYTES                        // ring of 4 slots
#define SLAB_HALFS (4 * SLOT_HALFS)               // 6912
#define ZST_OFF    (SLAB_OFF + SLAB_HALFS * 2)    // 56832, tanh(Z) stash
#define ST_HALFS   (4 * D_DIM * 32 * 8)           // 31744 per stash
#define FST_OFF    (ZST_OFF + ST_HALFS * 2)       // 120320, sigmoid(F2) stash
#define SMEM_BYTES (FST_OFF + ST_HALFS * 2)       // 183808

#if defined(__has_builtin)
#  if __has_builtin(__builtin_amdgcn_global_load_async_to_lds_b128)
#    define HAS_ASYNC 1
#  endif
#endif
#ifndef HAS_ASYNC
#  define HAS_ASYNC 0
#endif

__device__ __forceinline__ float fast_sig(float x)  { return 1.0f / (1.0f + __expf(-x)); }
__device__ __forceinline__ float fast_tanh(float x) { return 2.0f / (1.0f + __expf(-2.0f * x)) - 1.0f; }

__device__ __forceinline__ void wait_async0() {
#if HAS_ASYNC
#  if __has_builtin(__builtin_amdgcn_s_wait_asynccnt)
    __builtin_amdgcn_s_wait_asynccnt(0);
#  else
    asm volatile("s_wait_asynccnt 0x0" ::: "memory");
#  endif
#endif
}

#if HAS_ASYNC
typedef __attribute__((address_space(1))) v4i g_v4i;   // global int4
typedef __attribute__((address_space(3))) v4i l_v4i;   // LDS int4
__device__ __forceinline__ void async_cp16(const void* g, void* l) {
    // (global src, lds dst, imm offset, imm cpol); LDS generic addr low 32 bits = LDS byte offset
    __builtin_amdgcn_global_load_async_to_lds_b128(
        (g_v4i*)(uintptr_t)g, (l_v4i*)(uint32_t)(uintptr_t)l, 0, 0);
}
#endif

// ---------------- weight pre-swizzle: W[48][16][3][3][3] f32 -> WMMA A fragments f16 ------------
__global__ void prep_weights(const float* __restrict__ Wg, _Float16* __restrict__ A) {
    int idx = blockIdx.x * 256 + threadIdx.x;
    if (idx >= A_HALFS) return;
    int e    = idx & 15;
    int lane = (idx >> 4) & 31;
    int c    = (idx >> 9) % KCH;
    int mt   = idx / (KCH * 512);
    int half = lane >> 4, row = lane & 15;
    int koff = 16 * half + e;
    int k    = 32 * c + koff;
    int tap  = k >> 4, cin = k & 15;
    float val = 0.0f;
    if (tap < 27) {
        int kd = tap / 9, kh = (tap % 9) / 3, kw = tap % 3;
        int ch = 16 * mt + row;
        val = Wg[(((ch * 16 + cin) * 3 + kd) * 3 + kh) * 3 + kw];
    }
    A[idx] = (_Float16)val;
}

#if HAS_ASYNC
// ---------------- x pre-pass: padded channel-last f16 copy, transposed via LDS ------------------
__global__ __launch_bounds__(256) void prep_x(const float* __restrict__ x, _Float16* __restrict__ xt) {
    __shared__ _Float16 lh[16 * 272];          // padded stride to spread LDS banks
    int hz  = blockIdx.x;                      // 0..257
    int dz  = blockIdx.y;                      // 0..32
    int b   = blockIdx.z;
    int tid = threadIdx.x;
    size_t rowbase = (((size_t)b * DP + dz) * HP + hz) * (size_t)WP * 16;
    bool border = (dz == 0) | (dz == DP - 1) | (hz == 0) | (hz == HP - 1);
    if (!border) {
        for (int i = tid; i < 4096; i += 256) {
            int cin = i >> 8, w = i & 255;     // coalesced 256-wide reads per cin
            float v = x[((size_t)(b * 16 + cin) * D_DIM + (dz - 1)) * (size_t)(H_DIM * W_DIM)
                        + (size_t)(hz - 1) * W_DIM + w];
            lh[cin * 272 + w] = (_Float16)v;
        }
    }
    __syncthreads();
    for (int i = tid; i < WP * 16; i += 256) { // coalesced channel-last writes
        int wz = i >> 4, cin = i & 15;
        _Float16 v = (_Float16)0.0f;
        if (!border && wz >= 1 && wz <= 256) v = lh[cin * 272 + (wz - 1)];
        xt[rowbase + i] = v;
    }
}

// Stage depth-slice s into slab slot (s & 3): 6 rows x 576 contiguous bytes, pure async copies.
__device__ __forceinline__ void stage_async(const _Float16* __restrict__ xt, _Float16* slab,
                                            int b, int s, int h0, int w0, int tid) {
    int slotBase = (s & 3) * SLOT_HALFS;
    size_t gbase = (((size_t)b * DP + (s + 1)) * HP + h0) * (size_t)WP * 16 + (size_t)w0 * 16;
    for (int i = tid; i < 216; i += 128) {     // 216 x 16B chunks
        int r  = i / 36;
        int ch = i - r * 36;
        const _Float16* g = xt + gbase + (size_t)r * (WP * 16) + ch * 8;
        _Float16*       l = slab + slotBase + r * (18 * 16) + ch * 8;
        async_cp16(g, l);
    }
}
#else
// Fallback synchronous staging from original f32 x (with SAME-padding checks).
__device__ __forceinline__ void stage_sync(const float* __restrict__ x, _Float16* slab,
                                           int b, int s, int h0, int w0, int tid) {
    int  slotBase = (s & 3) * SLOT_HALFS;
    bool valid = (s >= 0) && (s < D_DIM);
    for (int i = tid; i < 16 * 6 * 18; i += 128) {
        int w = i % 18, h = (i / 18) % 6, cin = i / 108;
        int gh = h0 - 1 + h, gw = w0 - 1 + w;
        float v = 0.0f;
        if (valid && (unsigned)gh < H_DIM && (unsigned)gw < W_DIM)
            v = x[((size_t)(b * 16 + cin) * D_DIM + s) * (size_t)(H_DIM * W_DIM)
                  + (size_t)gh * W_DIM + gw];
        slab[slotBase + (h * 18 + w) * 16 + cin] = (_Float16)v;
    }
}
#endif

// pick slab slot base for tap depth offset kd (0..3), kd is a compile-time literal after unroll
#define SLOTBASE(kd) ((kd) == 0 ? slm1 : (kd) == 1 ? sl0 : (kd) == 2 ? slp1 : slp2)

__global__ __launch_bounds__(128) void biqrnn3d_kernel(
        const float* __restrict__ x, const _Float16* __restrict__ xt,
        const float* __restrict__ bias, const _Float16* __restrict__ Aglob,
        float* __restrict__ out) {
    extern __shared__ __align__(32) char smem[];
    _Float16* Alds = (_Float16*)smem;
    _Float16* slab = (_Float16*)(smem + SLAB_OFF);
    _Float16* zst  = (_Float16*)(smem + ZST_OFF);
    _Float16* fst  = (_Float16*)(smem + FST_OFF);

    const int tid  = threadIdx.x;
    const int wv   = tid >> 5, lane = tid & 31;
    const int half = lane >> 4, N = lane & 15;
    const int w0   = blockIdx.x * 16;
    const int h0   = blockIdx.y * 4;
    const int b    = blockIdx.z;
    const int hrow = h0 + wv;

    // A matrix -> LDS (block cooperative)
    {
        const int* src = (const int*)Aglob;
        int*       dst = (int*)Alds;
        for (int i = tid; i < A_BYTES / 4; i += 128) dst[i] = src[i];
    }

    float bz[8], b1[8], b2[8];
#pragma unroll
    for (int v = 0; v < 8; ++v) {
        int ch = v + 8 * half;
        bz[v] = bias[ch];
        b1[v] = bias[16 + ch];
        b2[v] = bias[32 + ch];
    }

#if HAS_ASYNC
    stage_async(xt, slab, b, -1, h0, w0, tid);
    stage_async(xt, slab, b,  0, h0, w0, tid);
    stage_async(xt, slab, b,  1, h0, w0, tid);
    wait_async0();
#else
    stage_sync(x, slab, b, -1, h0, w0, tid);
    stage_sync(x, slab, b,  0, h0, w0, tid);
    stage_sync(x, slab, b,  1, h0, w0, tid);
#endif
    __syncthreads();

    const size_t HW = (size_t)H_DIM * W_DIM;
    const int colOff = (wv * 18 + N) * 16;          // per-lane B column offset (halfs)
    const _Float16* aLane = Alds + lane * 16;       // per-lane A fragment base

    // ---------------- forward: WMMA conv + gates + forward scan ----------------
    v8f hprev = {};
    for (int d = 0; d < D_DIM; ++d) {
        // prefetch slice d+2 into slot (d+2)&3 (overlaps with this depth's compute)
        if (d + 2 <= D_DIM) {
#if HAS_ASYNC
            stage_async(xt, slab, b, d + 2, h0, w0, tid);
#else
            stage_sync(x, slab, b, d + 2, h0, w0, tid);
#endif
        }

        const int slm1 = ((d - 1) & 3) * SLOT_HALFS;   // slice d-1
        const int sl0  = ((d    ) & 3) * SLOT_HALFS;   // slice d
        const int slp1 = ((d + 1) & 3) * SLOT_HALFS;   // slice d+1
        const int slp2 = ((d + 2) & 3) * SLOT_HALFS;   // pad tap (multiplied by zero weights)

        v8f accZ = {}, accF1 = {}, accF2 = {};
        // software-pipelined fragment loads
        v16h bf, a0, a1, a2;
        {
            int addr = SLOTBASE(0) + colOff;           // chunk 0: taps 0/1
            int tapoff = half ? (0 * 18 + 1) * 16 : 0; // tap0=(0,0,0), tap1=(0,0,1)
            bf = *(const v16h*)(slab + addr + tapoff);
            a0 = *(const v16h*)(aLane + (0 * KCH + 0) * 512);
            a1 = *(const v16h*)(aLane + (1 * KCH + 0) * 512);
            a2 = *(const v16h*)(aLane + (2 * KCH + 0) * 512);
        }
#pragma unroll
        for (int c = 0; c < KCH; ++c) {
            v16h bfn, a0n, a1n, a2n;
            if (c < KCH - 1) {
                const int cn   = c + 1;
                const int tapA = 2 * cn, tapB = 2 * cn + 1;
                const int kdA = tapA / 9, khA = (tapA % 9) / 3, kwA = tapA % 3;
                const int kdB = tapB / 9, khB = (tapB % 9) / 3, kwB = tapB % 3;
                int base   = half ? SLOTBASE(kdB) : SLOTBASE(kdA);
                int tapoff = half ? (khB * 18 + kwB) * 16 : (khA * 18 + kwA) * 16;
                bfn = *(const v16h*)(slab + base + tapoff + colOff);
                a0n = *(const v16h*)(aLane + (0 * KCH + cn) * 512);
                a1n = *(const v16h*)(aLane + (1 * KCH + cn) * 512);
                a2n = *(const v16h*)(aLane + (2 * KCH + cn) * 512);
            }
            accZ  = __builtin_amdgcn_wmma_f32_16x16x32_f16(false, a0, false, bf, (short)0, accZ,  false, false);
            accF1 = __builtin_amdgcn_wmma_f32_16x16x32_f16(false, a1, false, bf, (short)0, accF1, false, false);
            accF2 = __builtin_amdgcn_wmma_f32_16x16x32_f16(false, a2, false, bf, (short)0, accF2, false, false);
            if (c < KCH - 1) { bf = bfn; a0 = a0n; a1 = a1n; a2 = a2n; }
        }

        v8h zh, fh;
#pragma unroll
        for (int v = 0; v < 8; ++v) {
            float z  = fast_tanh(accZ[v]  + bz[v]);
            float f1 = fast_sig (accF1[v] + b1[v]);
            float f2 = fast_sig (accF2[v] + b2[v]);
            hprev[v] = f1 * hprev[v] + (1.0f - f1) * z;
            zh[v] = (_Float16)z;
            fh[v] = (_Float16)f2;
            int ch = v + 8 * half;
            size_t oi = ((size_t)(b * 16 + ch) * D_DIM + d) * HW + (size_t)hrow * W_DIM + w0 + N;
            out[oi] = hprev[v];
        }
        *(v8h*)(zst + ((wv * D_DIM + d) * 32 + lane) * 8) = zh;
        *(v8h*)(fst + ((wv * D_DIM + d) * 32 + lane) * 8) = fh;

        wait_async0();        // staged slice d+2 resident before next iteration's readers
        __syncthreads();
    }

    // ---------------- backward: replay from LDS, accumulate into out ----------------
    v8f hb = {};
    for (int d = D_DIM - 1; d >= 0; --d) {
        v8h zh = *(const v8h*)(zst + ((wv * D_DIM + d) * 32 + lane) * 8);
        v8h fh = *(const v8h*)(fst + ((wv * D_DIM + d) * 32 + lane) * 8);
#pragma unroll
        for (int v = 0; v < 8; ++v) {
            float z  = (float)zh[v];
            float f2 = (float)fh[v];
            hb[v] = f2 * hb[v] + (1.0f - f2) * z;
            int ch = v + 8 * half;
            size_t oi = ((size_t)(b * 16 + ch) * D_DIM + d) * HW + (size_t)hrow * W_DIM + w0 + N;
            out[oi] += hb[v];
        }
    }
}

extern "C" void kernel_launch(void* const* d_in, const int* in_sizes, int n_in,
                              void* d_out, int out_size, void* d_ws, size_t ws_size,
                              hipStream_t stream) {
    (void)in_sizes; (void)n_in; (void)out_size; (void)ws_size;
    const float* x    = (const float*)d_in[0];
    const float* W    = (const float*)d_in[1];
    const float* bias = (const float*)d_in[2];
    float*       out  = (float*)d_out;
    _Float16*    A    = (_Float16*)d_ws;
    _Float16*    xt   = (_Float16*)((char*)d_ws + XT_OFF_BYTES);

    prep_weights<<<(A_HALFS + 255) / 256, 256, 0, stream>>>(W, A);
#if HAS_ASYNC
    {
        dim3 pg(HP, DP, BATCH);
        prep_x<<<pg, 256, 0, stream>>>(x, xt);
    }
#endif

    (void)hipFuncSetAttribute((const void*)biqrnn3d_kernel,
                              hipFuncAttributeMaxDynamicSharedMemorySize, SMEM_BYTES);

    dim3 grid(W_DIM / 16, H_DIM / 4, BATCH);
    biqrnn3d_kernel<<<grid, 128, SMEM_BYTES, stream>>>(x, xt, bias, A, out);
}
